// CriticMAAC_62981400428853
// MI455X (gfx1250) — compile-verified, matrix operand order checked by reading, not run on hardware
//
#include <hip/hip_runtime.h>

// ---------------------------------------------------------------------------
// MAAC critic forward, CDNA5 (gfx1250) WMMA bf16 pipeline.
// A=8 agents, B=8192 batch, OBS=128, ACT=32, H=128, K=4 heads.
// ---------------------------------------------------------------------------

#define DEVFN __device__ __forceinline__

constexpr int AGENTS = 8;
constexpr int BATCH  = 8192;
constexpr int OBS_D  = 128;
constexpr int ACT_D  = 32;
constexpr int HID    = 128;
constexpr int HEADS  = 4;

typedef __attribute__((ext_vector_type(16))) __bf16 v16bf;
typedef __attribute__((ext_vector_type(8)))  float  v8f;

union Frag {
  v16bf v;
  unsigned int u[8];
  unsigned short s[16];
};

DEVFN unsigned short f32_to_bf16(float f) {
  union { float f; unsigned int u; } x; x.f = f;
  unsigned int u = x.u;
  unsigned int r = u + 0x7FFFu + ((u >> 16) & 1u);   // round-to-nearest-even
  return (unsigned short)(r >> 16);
}
DEVFN float bf16_to_f32(unsigned short s) {
  union { float f; unsigned int u; } x; x.u = ((unsigned int)s) << 16;
  return x.f;
}
// unpack a u32 holding two bf16 -> two f32 (shift-only, no cvt needed)
DEVFN void bf16x2_unpack(unsigned int u, float& lo, float& hi) {
  union { float f; unsigned int v; } a, b;
  a.v = u << 16;
  b.v = u & 0xFFFF0000u;
  lo = a.f; hi = b.f;
}
DEVFN unsigned int bf16x2_pack(float lo, float hi) {
  return (unsigned int)f32_to_bf16(lo) | ((unsigned int)f32_to_bf16(hi) << 16);
}
DEVFN v8f zero_v8f() {
  v8f z = {0.f, 0.f, 0.f, 0.f, 0.f, 0.f, 0.f, 0.f};
  return z;
}

DEVFN v8f wmma_bf16(const Frag& a, const Frag& b, v8f c) {
  // D = A(16x32 bf16) * B(32x16 bf16) + C(16x16 f32)
  return __builtin_amdgcn_wmma_f32_16x16x32_bf16(
      /*neg_a=*/false, a.v, /*neg_b=*/false, b.v,
      /*c_mod=*/(short)0, c, /*reuse_a=*/false, /*reuse_b=*/false);
}

// A/B fragment from bf16 row-major [rows, stride] buffer (global or LDS).
// 16-bit A layout: lane&15 = row(M) / col(N); lane>>4 picks K {0..7} vs {8..15},
// second 16B load picks K {16..23} vs {24..31}.
DEVFN Frag frag_ld_bf16(const unsigned short* base, int stride, int m0, int k0, int lane) {
  Frag f;
  const int row = m0 + (lane & 15);
  const int kb  = k0 + ((lane >> 4) << 3);
  const unsigned short* p = base + (size_t)row * stride + kb;
  const uint4 a = *(const uint4*)(p);
  const uint4 b = *(const uint4*)(p + 16);
  f.u[0] = a.x; f.u[1] = a.y; f.u[2] = a.z; f.u[3] = a.w;
  f.u[4] = b.x; f.u[5] = b.y; f.u[6] = b.z; f.u[7] = b.w;
  return f;
}

// A fragment from f32 row-major buffer, converted to bf16 on the fly.
DEVFN Frag frag_ld_f32(const float* base, int stride, int m0, int k0, int lane) {
  Frag f;
  const int row = m0 + (lane & 15);
  const int kb  = k0 + ((lane >> 4) << 3);
  const float* p = base + (size_t)row * stride + kb;
  float4 x0 = *(const float4*)(p + 0);
  float4 x1 = *(const float4*)(p + 4);
  float4 y0 = *(const float4*)(p + 16);
  float4 y1 = *(const float4*)(p + 20);
  f.s[0]  = f32_to_bf16(x0.x); f.s[1]  = f32_to_bf16(x0.y);
  f.s[2]  = f32_to_bf16(x0.z); f.s[3]  = f32_to_bf16(x0.w);
  f.s[4]  = f32_to_bf16(x1.x); f.s[5]  = f32_to_bf16(x1.y);
  f.s[6]  = f32_to_bf16(x1.z); f.s[7]  = f32_to_bf16(x1.w);
  f.s[8]  = f32_to_bf16(y0.x); f.s[9]  = f32_to_bf16(y0.y);
  f.s[10] = f32_to_bf16(y0.z); f.s[11] = f32_to_bf16(y0.w);
  f.s[12] = f32_to_bf16(y1.x); f.s[13] = f32_to_bf16(y1.y);
  f.s[14] = f32_to_bf16(y1.z); f.s[15] = f32_to_bf16(y1.w);
  return f;
}

// Scatter one 16x16 f32 accumulator tile (bias + activation) as bf16.
// C/D layout: VGPR r, lanes 0-15 -> M=r,N=lane; lanes 16-31 -> M=8+r,N=lane-16.
// act: 0 = none, 1 = relu, 2 = leaky_relu(0.01)
DEVFN void store_tile_bf16(unsigned short* dst, int stride, int m0, int n0,
                           int lane, v8f acc, float bias, int act) {
  const int col   = n0 + (lane & 15);
  const int rbase = m0 + ((lane >> 4) << 3);
#pragma unroll
  for (int r = 0; r < 8; ++r) {
    float vv = acc[r] + bias;
    if (act == 1)      vv = fmaxf(vv, 0.0f);
    else if (act == 2) vv = (vv > 0.0f) ? vv : 0.01f * vv;
    dst[(size_t)(rbase + r) * stride + col] = f32_to_bf16(vv);
  }
}

// Spread a prefetch of [bytes] across the 256 threads of a block
// (lowers to global_prefetch_b8; speculative, no counters touched).
DEVFN void block_prefetch(const void* p, int bytes) {
  const char* c = (const char*)p;
  for (int off = (int)threadIdx.x * 128; off < bytes; off += 256 * 128)
    __builtin_prefetch(c + off, 0, 1);
}

// ---------------------------------------------------------------------------
// Kernel P: weight transpose + f32 -> bf16.  src [N,R,C] -> dst [N,C,R] bf16.
// ---------------------------------------------------------------------------
__global__ __launch_bounds__(256) void transpose_cvt_kernel(
    const float* __restrict__ src, unsigned short* __restrict__ dst,
    int N, int R, int C) {
  const int total = N * R * C;
  for (int idx = blockIdx.x * blockDim.x + threadIdx.x; idx < total;
       idx += gridDim.x * blockDim.x) {
    int n = idx / (R * C);
    int rem = idx - n * R * C;
    int r = rem / C;
    int c = rem - r * C;
    dst[((size_t)n * C + c) * R + r] = f32_to_bf16(src[idx]);
  }
}

// ---------------------------------------------------------------------------
// Kernel 1: per-agent encoders  e = relu([relu(obs*Wo+b)|relu(act*Wa+b)]*Woa+b)
// grid = (B/128, A); 8 waves, each owns a 16-row x 128-col strip.
// dynamic LDS: 128 x 256 bf16 oa tile (64 KB).
// ---------------------------------------------------------------------------
__global__ __launch_bounds__(256) void encoder_kernel(
    const float* __restrict__ obs, const float* __restrict__ act,
    const unsigned short* __restrict__ Wt_obs, const float* __restrict__ b_obs,
    const unsigned short* __restrict__ Wt_act, const float* __restrict__ b_act,
    const unsigned short* __restrict__ Wt_oa,  const float* __restrict__ b_oa,
    unsigned short* __restrict__ e_out) {
  extern __shared__ unsigned short lds[];              // [128][256]
  const int a     = blockIdx.y;
  const int btile = blockIdx.x * 128;
  const int lane  = threadIdx.x & 31;
  const int wave  = threadIdx.x >> 5;
  const int m0    = wave * 16;

  const float* obsA = obs + (size_t)a * BATCH * OBS_D;
  const float* actA = act + (size_t)a * BATCH * ACT_D;
  const unsigned short* WoT = Wt_obs + (size_t)a * HID * OBS_D;
  const unsigned short* WaT = Wt_act + (size_t)a * HID * ACT_D;
  const unsigned short* WcT = Wt_oa  + (size_t)a * HID * (2 * HID);

  // pull this agent's weights toward the caches before the wmma burst
  block_prefetch(WoT, HID * OBS_D * 2);
  block_prefetch(WcT, HID * 2 * HID * 2);

  v8f acc[8];

  // ---- e_obs = relu(obs @ W_obs + b_obs) -> LDS cols [0,128)
#pragma unroll
  for (int i = 0; i < 8; ++i) acc[i] = zero_v8f();
  for (int kc = 0; kc < OBS_D; kc += 32) {
    Frag af = frag_ld_f32(obsA, OBS_D, btile + m0, kc, lane);
#pragma unroll
    for (int nt = 0; nt < 8; ++nt) {
      Frag bf = frag_ld_bf16(WoT, OBS_D, nt * 16, kc, lane);
      acc[nt] = wmma_bf16(af, bf, acc[nt]);
    }
  }
#pragma unroll
  for (int nt = 0; nt < 8; ++nt) {
    float bias = b_obs[a * HID + nt * 16 + (lane & 15)];
    store_tile_bf16(lds, 2 * HID, m0, nt * 16, lane, acc[nt], bias, 1);
  }

  // ---- e_act = relu(act @ W_act + b_act) -> LDS cols [128,256)
#pragma unroll
  for (int i = 0; i < 8; ++i) acc[i] = zero_v8f();
  {
    Frag af = frag_ld_f32(actA, ACT_D, btile + m0, 0, lane);
#pragma unroll
    for (int nt = 0; nt < 8; ++nt) {
      Frag bf = frag_ld_bf16(WaT, ACT_D, nt * 16, 0, lane);
      acc[nt] = wmma_bf16(af, bf, acc[nt]);
    }
  }
#pragma unroll
  for (int nt = 0; nt < 8; ++nt) {
    float bias = b_act[a * HID + nt * 16 + (lane & 15)];
    store_tile_bf16(lds, 2 * HID, m0, HID + nt * 16, lane, acc[nt], bias, 1);
  }

  __syncthreads();

  // ---- e = relu(oa @ W_oa + b_oa) -> global bf16
#pragma unroll
  for (int i = 0; i < 8; ++i) acc[i] = zero_v8f();
  for (int kc = 0; kc < 2 * HID; kc += 32) {
    Frag af = frag_ld_bf16(lds, 2 * HID, m0, kc, lane);
#pragma unroll
    for (int nt = 0; nt < 8; ++nt) {
      Frag bf = frag_ld_bf16(WcT, 2 * HID, nt * 16, kc, lane);
      acc[nt] = wmma_bf16(af, bf, acc[nt]);
    }
  }
  unsigned short* eA = e_out + (size_t)a * BATCH * HID;
#pragma unroll
  for (int nt = 0; nt < 8; ++nt) {
    float bias = b_oa[a * HID + nt * 16 + (lane & 15)];
    store_tile_bf16(eA, HID, btile + m0, nt * 16, lane, acc[nt], bias, 1);
  }
}

// ---------------------------------------------------------------------------
// Kernel 2: fused QKV projections (WMMA) + 8x8 masked softmax attention.
// grid = (B/32, K heads); q/k/v tiles for all 8 agents kept in LDS (192 KB).
// t-outer GEMM loop: weight B-fragments hoisted into registers once and
// reused across all 8 agents (8x fewer weight loads).
// Attention phase is fully vectorized: ds_load_b128 / global_store_b128.
// ---------------------------------------------------------------------------
__global__ __launch_bounds__(256) void qkv_attn_kernel(
    const unsigned short* __restrict__ e,
    const unsigned short* __restrict__ WqT, const float* __restrict__ b_q,
    const unsigned short* __restrict__ WkT, const float* __restrict__ b_k,
    const unsigned short* __restrict__ WvT, const float* __restrict__ b_v,
    unsigned short* __restrict__ x_out) {
  extern __shared__ unsigned short lds2[];
  unsigned short* qs  = lds2;                                  // [8][32][128]
  unsigned short* ks_ = lds2 + AGENTS * 32 * HID;
  unsigned short* vs  = lds2 + 2 * AGENTS * 32 * HID;

  const int kk    = blockIdx.y;
  const int btile = blockIdx.x * 32;
  const int lane  = threadIdx.x & 31;
  const int wave  = threadIdx.x >> 5;
  const int mt    = wave & 1;      // 16-row sub-tile
  const int nq    = wave >> 1;     // 32-col quarter

  block_prefetch(WqT + (size_t)kk * HID * HID, HID * HID * 2);
  block_prefetch(WkT + (size_t)kk * HID * HID, HID * HID * 2);
  block_prefetch(WvT + (size_t)kk * HID * HID, HID * HID * 2);

#pragma unroll
  for (int t = 0; t < 3; ++t) {
    const unsigned short* W  = (t == 0) ? WqT : (t == 1) ? WkT : WvT;
    const float*          bb = (t == 0) ? b_q : (t == 1) ? b_k : b_v;
    unsigned short*      dst = (t == 0) ? qs  : (t == 1) ? ks_ : vs;
    const unsigned short* Wh = W + (size_t)kk * HID * HID;

    // hoist this wave's weight fragments (shared by all agents)
    Frag bf0[4], bf1[4];
#pragma unroll
    for (int kc4 = 0; kc4 < 4; ++kc4) {
      bf0[kc4] = frag_ld_bf16(Wh, HID, nq * 32,      kc4 * 32, lane);
      bf1[kc4] = frag_ld_bf16(Wh, HID, nq * 32 + 16, kc4 * 32, lane);
    }
    const float bias0 = bb[kk * HID + nq * 32 + (lane & 15)];
    const float bias1 = bb[kk * HID + nq * 32 + 16 + (lane & 15)];
    const int   actm  = (t == 2) ? 2 : 0;      // leaky_relu only on v

    for (int a = 0; a < AGENTS; ++a) {
      const unsigned short* eA = e + (size_t)a * BATCH * HID;
      v8f acc0 = zero_v8f(), acc1 = zero_v8f();
#pragma unroll
      for (int kc4 = 0; kc4 < 4; ++kc4) {
        Frag af = frag_ld_bf16(eA, HID, btile + mt * 16, kc4 * 32, lane);
        acc0 = wmma_bf16(af, bf0[kc4], acc0);
        acc1 = wmma_bf16(af, bf1[kc4], acc1);
      }
      store_tile_bf16(dst + a * 32 * HID, HID, mt * 16, nq * 32,      lane, acc0, bias0, actm);
      store_tile_bf16(dst + a * 32 * HID, HID, mt * 16, nq * 32 + 16, lane, acc1, bias1, actm);
    }
  }
  __syncthreads();

  // ---- attention: one thread per (agent i, batch row b); vectorized LDS IO
  const int i    = wave;   // 0..7
  const int brow = lane;   // 0..31
  const unsigned short* qrow = qs + (i * 32 + brow) * HID;
  const float scale = 0.3779644730092272f;   // 1/sqrt(A-1)

  float logit[AGENTS];
#pragma unroll
  for (int j = 0; j < AGENTS; ++j) logit[j] = 0.0f;

  for (int hc = 0; hc < HID; hc += 8) {
    const uint4 qv = *(const uint4*)(qrow + hc);
    float qf[8];
    bf16x2_unpack(qv.x, qf[0], qf[1]);
    bf16x2_unpack(qv.y, qf[2], qf[3]);
    bf16x2_unpack(qv.z, qf[4], qf[5]);
    bf16x2_unpack(qv.w, qf[6], qf[7]);
#pragma unroll
    for (int j = 0; j < AGENTS; ++j) {
      const uint4 kv = *(const uint4*)(ks_ + (j * 32 + brow) * HID + hc);
      float kf[8];
      bf16x2_unpack(kv.x, kf[0], kf[1]);
      bf16x2_unpack(kv.y, kf[2], kf[3]);
      bf16x2_unpack(kv.z, kf[4], kf[5]);
      bf16x2_unpack(kv.w, kf[6], kf[7]);
      float d = logit[j];
#pragma unroll
      for (int q8 = 0; q8 < 8; ++q8) d = fmaf(qf[q8], kf[q8], d);
      logit[j] = d;
    }
  }
#pragma unroll
  for (int j = 0; j < AGENTS; ++j)
    logit[j] = (j == i) ? -1e9f : logit[j] * scale;

  float mx = -1e30f;
#pragma unroll
  for (int j = 0; j < AGENTS; ++j) mx = fmaxf(mx, logit[j]);
  float wsum = 0.0f, w[AGENTS];
#pragma unroll
  for (int j = 0; j < AGENTS; ++j) { w[j] = __expf(logit[j] - mx); wsum += w[j]; }
  const float inv = 1.0f / wsum;
#pragma unroll
  for (int j = 0; j < AGENTS; ++j) w[j] *= inv;

  unsigned short* xrow = x_out +
      ((size_t)(kk * AGENTS + i) * BATCH + (btile + brow)) * HID;
  for (int hc = 0; hc < HID; hc += 8) {
    float s[8];
#pragma unroll
    for (int q8 = 0; q8 < 8; ++q8) s[q8] = 0.0f;
#pragma unroll
    for (int j = 0; j < AGENTS; ++j) {
      const uint4 vv = *(const uint4*)(vs + (j * 32 + brow) * HID + hc);
      const float wj = w[j];
      float vf[8];
      bf16x2_unpack(vv.x, vf[0], vf[1]);
      bf16x2_unpack(vv.y, vf[2], vf[3]);
      bf16x2_unpack(vv.z, vf[4], vf[5]);
      bf16x2_unpack(vv.w, vf[6], vf[7]);
#pragma unroll
      for (int q8 = 0; q8 < 8; ++q8) s[q8] = fmaf(wj, vf[q8], s[q8]);
    }
    uint4 o;
    o.x = bf16x2_pack(s[0], s[1]);
    o.y = bf16x2_pack(s[2], s[3]);
    o.z = bf16x2_pack(s[4], s[5]);
    o.w = bf16x2_pack(s[6], s[7]);
    *(uint4*)(xrow + hc) = o;
  }
}

// ---------------------------------------------------------------------------
// Kernel 3: out = relu([e|x0|x1|x2|x3] @ W_ex + b_ex); qvals = out @ W_qval + b
// grid = (B/128, A); 8 waves as in encoder. LDS: 128x128 bf16 out tile (32 KB).
// ---------------------------------------------------------------------------
__global__ __launch_bounds__(256) void head_kernel(
    const unsigned short* __restrict__ e,
    const unsigned short* __restrict__ x,
    const unsigned short* __restrict__ WexT, const float* __restrict__ b_ex,
    const float* __restrict__ W_qval, const float* __restrict__ b_qval,
    float* __restrict__ out) {
  extern __shared__ unsigned short lds3[];             // [128][128]
  const int a     = blockIdx.y;
  const int btile = blockIdx.x * 128;
  const int lane  = threadIdx.x & 31;
  const int wave  = threadIdx.x >> 5;
  const int m0    = wave * 16;

  const unsigned short* WeT = WexT + (size_t)a * HID * (5 * HID);
  block_prefetch(WeT, HID * 5 * HID * 2);

  v8f acc[8];
#pragma unroll
  for (int i = 0; i < 8; ++i) acc[i] = zero_v8f();

  for (int c = 0; c < 20; ++c) {                       // 20 chunks of K=32 over 640
    const unsigned short* src;
    int koff;
    if (c < 4) {                                       // e section
      src  = e + (size_t)a * BATCH * HID;
      koff = c * 32;
    } else {                                           // head (c-4)>>2 of x
      const int hh = (c - 4) >> 2;
      src  = x + (size_t)(hh * AGENTS + a) * BATCH * HID;
      koff = ((c - 4) & 3) * 32;
    }
    Frag af = frag_ld_bf16(src, HID, btile + m0, koff, lane);
#pragma unroll
    for (int nt = 0; nt < 8; ++nt) {
      Frag bf = frag_ld_bf16(WeT, 5 * HID, nt * 16, c * 32, lane);
      acc[nt] = wmma_bf16(af, bf, acc[nt]);
    }
  }
#pragma unroll
  for (int nt = 0; nt < 8; ++nt) {
    float bias = b_ex[a * HID + nt * 16 + (lane & 15)];
    store_tile_bf16(lds3, HID, m0, nt * 16, lane, acc[nt], bias, 1);
  }
  __syncthreads();

  if (threadIdx.x < 128) {
    const int row = threadIdx.x;
    float s = b_qval[a];
    const unsigned short* r = lds3 + row * HID;
    for (int hc = 0; hc < HID; hc += 8) {
      const uint4 ov = *(const uint4*)(r + hc);
      float of[8];
      bf16x2_unpack(ov.x, of[0], of[1]);
      bf16x2_unpack(ov.y, of[2], of[3]);
      bf16x2_unpack(ov.z, of[4], of[5]);
      bf16x2_unpack(ov.w, of[6], of[7]);
      const float4 w0 = *(const float4*)(W_qval + a * HID + hc);
      const float4 w1 = *(const float4*)(W_qval + a * HID + hc + 4);
      s = fmaf(of[0], w0.x, s); s = fmaf(of[1], w0.y, s);
      s = fmaf(of[2], w0.z, s); s = fmaf(of[3], w0.w, s);
      s = fmaf(of[4], w1.x, s); s = fmaf(of[5], w1.y, s);
      s = fmaf(of[6], w1.z, s); s = fmaf(of[7], w1.w, s);
    }
    out[(size_t)a * BATCH + btile + row] = s;
  }
}

// ---------------------------------------------------------------------------
// Host launcher
// ---------------------------------------------------------------------------
extern "C" void kernel_launch(void* const* d_in, const int* in_sizes, int n_in,
                              void* d_out, int out_size, void* d_ws, size_t ws_size,
                              hipStream_t stream) {
  (void)in_sizes; (void)n_in; (void)out_size; (void)ws_size;

  const float* obs    = (const float*)d_in[0];
  const float* act    = (const float*)d_in[1];
  const float* W_obs  = (const float*)d_in[2];
  const float* b_obs  = (const float*)d_in[3];
  const float* W_act  = (const float*)d_in[4];
  const float* b_act  = (const float*)d_in[5];
  const float* W_oa   = (const float*)d_in[6];
  const float* b_oa   = (const float*)d_in[7];
  const float* W_ex   = (const float*)d_in[8];
  const float* b_ex   = (const float*)d_in[9];
  const float* W_qval = (const float*)d_in[10];
  const float* b_qval = (const float*)d_in[11];
  const float* W_q    = (const float*)d_in[12];
  const float* b_q    = (const float*)d_in[13];
  const float* W_k    = (const float*)d_in[14];
  const float* b_k    = (const float*)d_in[15];
  const float* W_v    = (const float*)d_in[16];
  const float* b_v    = (const float*)d_in[17];

  char* ws = (char*)d_ws;
  size_t off = 0;
  auto take = [&](size_t bytes) -> char* {
    char* p = ws + off;
    off = (off + bytes + 255) & ~(size_t)255;
    return p;
  };
  unsigned short* e_ws  = (unsigned short*)take((size_t)AGENTS * BATCH * HID * 2);
  unsigned short* x_ws  = (unsigned short*)take((size_t)HEADS * AGENTS * BATCH * HID * 2);
  unsigned short* WtObs = (unsigned short*)take((size_t)AGENTS * HID * OBS_D * 2);
  unsigned short* WtAct = (unsigned short*)take((size_t)AGENTS * HID * ACT_D * 2);
  unsigned short* WtOa  = (unsigned short*)take((size_t)AGENTS * HID * 2 * HID * 2);
  unsigned short* WtEx  = (unsigned short*)take((size_t)AGENTS * HID * 5 * HID * 2);
  unsigned short* WtQ   = (unsigned short*)take((size_t)HEADS * HID * HID * 2);
  unsigned short* WtK   = (unsigned short*)take((size_t)HEADS * HID * HID * 2);
  unsigned short* WtV   = (unsigned short*)take((size_t)HEADS * HID * HID * 2);

  const dim3 blk(256);

  // Weight prep: transpose to [out, in] + convert to bf16 (tiny, one-shot).
  transpose_cvt_kernel<<<dim3(256),  blk, 0, stream>>>(W_obs, WtObs, AGENTS, OBS_D,   HID);
  transpose_cvt_kernel<<<dim3(64),   blk, 0, stream>>>(W_act, WtAct, AGENTS, ACT_D,   HID);
  transpose_cvt_kernel<<<dim3(512),  blk, 0, stream>>>(W_oa,  WtOa,  AGENTS, 2 * HID, HID);
  transpose_cvt_kernel<<<dim3(1024), blk, 0, stream>>>(W_ex,  WtEx,  AGENTS, 5 * HID, HID);
  transpose_cvt_kernel<<<dim3(256),  blk, 0, stream>>>(W_q,   WtQ,   HEADS,  HID,     HID);
  transpose_cvt_kernel<<<dim3(256),  blk, 0, stream>>>(W_k,   WtK,   HEADS,  HID,     HID);
  transpose_cvt_kernel<<<dim3(256),  blk, 0, stream>>>(W_v,   WtV,   HEADS,  HID,     HID);

  // Stage 1: per-agent encoders -> e (bf16)
  encoder_kernel<<<dim3(BATCH / 128, AGENTS), blk,
                   128 * (2 * HID) * sizeof(unsigned short), stream>>>(
      obs, act, WtObs, b_obs, WtAct, b_act, WtOa, b_oa, e_ws);

  // Stage 2: QKV + masked softmax attention -> x (bf16)
  qkv_attn_kernel<<<dim3(BATCH / 32, HEADS), blk,
                    3 * AGENTS * 32 * HID * sizeof(unsigned short), stream>>>(
      e_ws, WtQ, b_q, WtK, b_k, WtV, b_v, x_ws);

  // Stage 3: mixer + q-value head -> d_out [A, B, 1] f32
  head_kernel<<<dim3(BATCH / 128, AGENTS), blk,
                128 * HID * sizeof(unsigned short), stream>>>(
      e_ws, x_ws, WtEx, b_ex, W_qval, b_qval, (float*)d_out);
}
